// SAGE_82291573392195
// MI455X (gfx1250) — matrix operand order repaired and checked by reading, not compiled, not run on hardware
//
#include <hip/hip_runtime.h>

// ---------------------------------------------------------------------------
// GraphSAGE 3-layer inference for MI455X (gfx1250, wave32).
// GEMMs: v_wmma_f32_16x16x32_bf16, bf16 operands staged via async
// global->LDS copies (double-buffered), fp32 accumulate.
// All fp32->bf16 conversion hoisted out of the GEMM into one-shot passes.
// ---------------------------------------------------------------------------

typedef __attribute__((ext_vector_type(16))) __bf16 v16bf;
typedef __attribute__((ext_vector_type(8)))  float  v8f;
typedef __attribute__((ext_vector_type(4)))  int    v4i;

union FragAB { v16bf v; uint4 q[2]; unsigned int u[8]; };
union FragC  { v8f   v; float f[8]; };

// ---------------- async global->LDS copy (gfx1250 path + fallback) ---------

#if __has_builtin(__builtin_amdgcn_global_load_async_to_lds_b128)
#define HAVE_ASYNC_LDS 1
#else
#define HAVE_ASYNC_LDS 0
#endif

__device__ __forceinline__ void copy16_g2l(void* lds, const void* g) {
#if HAVE_ASYNC_LDS
    __builtin_amdgcn_global_load_async_to_lds_b128(
        (__attribute__((address_space(1))) v4i*)(g),
        (__attribute__((address_space(3))) v4i*)(lds),
        0, 0);
#else
    *(uint4*)lds = *(const uint4*)g;
#endif
}

__device__ __forceinline__ void wait_async_copies() {
#if HAVE_ASYNC_LDS
#if __has_builtin(__builtin_amdgcn_s_wait_asynccnt)
    __builtin_amdgcn_s_wait_asynccnt(0);
#else
    asm volatile("s_wait_asynccnt 0" ::: "memory");
#endif
#endif
}

// ---------------------------- helper kernels -------------------------------

__global__ void zero_f32(float* __restrict__ p, long n) {
    long i = (long)blockIdx.x * blockDim.x + threadIdx.x;
    long stride = (long)gridDim.x * blockDim.x;
    for (; i < n; i += stride) p[i] = 0.0f;
}

__global__ void f32_to_bf16(const float* __restrict__ in, __bf16* __restrict__ out, long n) {
    long i = (long)blockIdx.x * blockDim.x + threadIdx.x;
    long stride = (long)gridDim.x * blockDim.x;
    for (; i < n; i += stride) out[i] = (__bf16)in[i];
}

// WT[n][k] = W[k][n] as bf16; rows n >= N zero-padded (Npad = gridDim.x)
__global__ void wt_transpose_bf16(const float* __restrict__ W, __bf16* __restrict__ WT,
                                  int K, int N) {
    const int n = blockIdx.x;
    __bf16* dst = WT + (long)n * K;
    if (n < N) {
        for (int k = threadIdx.x; k < K; k += blockDim.x)
            dst[k] = (__bf16)W[(long)k * N + n];
    } else {
        for (int k = threadIdx.x; k < K; k += blockDim.x)
            dst[k] = (__bf16)0.0f;
    }
}

// One block per edge (grid-stride): agg[dst] += h[src], deg[dst] += 1
__global__ void scatter_add_f32(const float* __restrict__ h,
                                const int* __restrict__ src, const int* __restrict__ dst,
                                float* __restrict__ agg, float* __restrict__ deg,
                                int E, int F) {
    const int F4 = F >> 2;
    for (int e = blockIdx.x; e < E; e += gridDim.x) {
        const float4* hr = (const float4*)(h + (long)src[e] * F);
        float*        ar = agg + (long)dst[e] * F;
        for (int f = threadIdx.x; f < F4; f += blockDim.x) {
            const float4 v = hr[f];
            atomicAdd(&ar[4 * f + 0], v.x);
            atomicAdd(&ar[4 * f + 1], v.y);
            atomicAdd(&ar[4 * f + 2], v.z);
            atomicAdd(&ar[4 * f + 3], v.w);
        }
        if (threadIdx.x == 0) atomicAdd(&deg[dst[e]], 1.0f);
    }
}

__global__ void scatter_add_bf16(const __bf16* __restrict__ h,
                                 const int* __restrict__ src, const int* __restrict__ dst,
                                 float* __restrict__ agg, float* __restrict__ deg,
                                 int E, int F) {
    for (int e = blockIdx.x; e < E; e += gridDim.x) {
        const __bf16* hr = h + (long)src[e] * F;
        float*        ar = agg + (long)dst[e] * F;
        for (int f = threadIdx.x; f < F; f += blockDim.x)
            atomicAdd(&ar[f], (float)hr[f]);
        if (threadIdx.x == 0) atomicAdd(&deg[dst[e]], 1.0f);
    }
}

// mean[r,:] = bf16( agg[r,:] / max(deg[r],1) ), one block per row
__global__ void normalize_mean_bf16(const float* __restrict__ agg,
                                    const float* __restrict__ deg,
                                    __bf16* __restrict__ mean, int F) {
    const int r = blockIdx.x;
    const float inv = 1.0f / fmaxf(deg[r], 1.0f);
    const float* ar = agg + (long)r * F;
    __bf16*      mr = mean + (long)r * F;
    for (int f = threadIdx.x; f < F; f += blockDim.x)
        mr[f] = (__bf16)(ar[f] * inv);
}

// ------------------------- fused SAGE WMMA GEMM ----------------------------
// C[M,N] = act( Aself*Wself + Amean*Wneigh + bias )
// Aself/Amean: bf16 [M,K] row-major.  WTself/WTneigh: bf16 [Npad,K] (W^T,
// zero-padded to Npad = BN multiple), so every async tile load is in-bounds.
// M % 64 == 0, K % 32 == 0. 256 threads = 8 wave32; tile 64x128, K-step 32,
// double-buffered LDS with async global->LDS copies.

#define BM 64
#define BN 128
#define BK 32
#define LDK 40   // LDS K-stride (bf16 elems): 80B rows, 16B aligned chunks

__global__ __launch_bounds__(256)
void sage_gemm_bf16_wmma(const __bf16* __restrict__ Aself,
                         const __bf16* __restrict__ Amean,
                         const __bf16* __restrict__ WTself,
                         const __bf16* __restrict__ WTneigh,
                         const float*  __restrict__ bias,
                         __bf16* __restrict__ OutBf,
                         float*  __restrict__ OutF32,
                         int M, int K, int N, int relu) {
    __shared__ __align__(16) __bf16 As[2][2][BM][LDK];  // [buf][stream] 20.0 KB
    __shared__ __align__(16) __bf16 Bs[2][2][BN][LDK];  // [buf][stream] 40.0 KB

    const int t    = threadIdx.x;
    const int lane = t & 31;
    const int wave = t >> 5;
    const int lr   = lane & 15;
    const int half = lane >> 4;

    const int mblock = blockIdx.y * BM;
    const int nblock = blockIdx.x * BN;

    const int mt   = wave & 3;       // wave's 16-row tile
    const int nh   = wave >> 2;      // wave's 64-col half
    const int mrow = mt * 16 + lr;

    // per-thread staging coordinates (16B granules)
    const int str  = t >> 2;         // 0..63
    const int sko  = (t & 3) * 8;    // 0,8,16,24 (bf16 elems)

    FragC acc[4];
#pragma unroll
    for (int j = 0; j < 4; ++j)
#pragma unroll
        for (int r = 0; r < 8; ++r) acc[j].f[r] = 0.0f;

    // ---- tile stage: A 64x32 (1 granule/thread/stream), W^T 128x32 (2) ----
    auto stage = [&](int buf, int k0) {
#pragma unroll
        for (int s = 0; s < 2; ++s) {
            const __bf16* Ap = s ? Amean : Aself;
            copy16_g2l(&As[buf][s][str][sko],
                       Ap + (long)(mblock + str) * K + k0 + sko);
        }
#pragma unroll
        for (int s = 0; s < 2; ++s) {
            const __bf16* Wp = s ? WTneigh : WTself;
#pragma unroll
            for (int i = 0; i < 2; ++i) {
                const int n = str + i * 64;
                copy16_g2l(&Bs[buf][s][n][sko],
                           Wp + (long)(nblock + n) * K + k0 + sko);
            }
        }
    };

    const int nK = K / BK;
    stage(0, 0);

    for (int kb = 0; kb < nK; ++kb) {
        const int cur = kb & 1;
        wait_async_copies();
        __syncthreads();
        if (kb + 1 < nK) stage(cur ^ 1, (kb + 1) * BK);

        // ---- A fragments: per lane, two contiguous 16B runs
        //      elems 0..7  -> K = e + 8*half        (offset 8*half)
        //      elems 8..15 -> K = 16 + e-8 + 8*half (offset 16 + 8*half)
        FragAB a[2];
#pragma unroll
        for (int s = 0; s < 2; ++s) {
            const __bf16* ap = &As[cur][s][mrow][0];
            a[s].q[0] = *(const uint4*)(ap + 8 * half);
            a[s].q[1] = *(const uint4*)(ap + 16 + 8 * half);
        }

#pragma unroll
        for (int j = 0; j < 4; ++j) {
            const int colb = nh * 64 + j * 16 + lr;   // B column (N)
#pragma unroll
            for (int s = 0; s < 2; ++s) {
                // B fragment: same K order as A: two 16B runs at 8*half, 16 + 8*half
                const __bf16* bp = &Bs[cur][s][colb][0];
                FragAB b;
                b.q[0] = *(const uint4*)(bp + 8 * half);
                b.q[1] = *(const uint4*)(bp + 16 + 8 * half);
                acc[j].v = __builtin_amdgcn_wmma_f32_16x16x32_bf16(
                    false, a[s].v, false, b.v, (short)0, acc[j].v, false, false);
            }
        }
    }

    // ---- epilogue: bias, optional ReLU; C layout: VGPR r, lanes 0-15 ->
    // (M=r, N=lane); lanes 16-31 -> (M=8+r, N=lane-16)
#pragma unroll
    for (int j = 0; j < 4; ++j) {
        const int col = nblock + nh * 64 + j * 16 + lr;
        if (col < N) {
            const float bv = bias[col];
#pragma unroll
            for (int r = 0; r < 8; ++r) {
                const int row = mblock + mt * 16 + r + 8 * half;
                float v = acc[j].f[r] + bv;
                if (relu) v = fmaxf(v, 0.0f);
                if (OutF32) OutF32[(long)row * N + col] = v;
                else        OutBf[(long)row * N + col] = (__bf16)v;
            }
        }
    }
}

// ------------------------------- launcher ----------------------------------

extern "C" void kernel_launch(void* const* d_in, const int* in_sizes, int n_in,
                              void* d_out, int out_size, void* d_ws, size_t ws_size,
                              hipStream_t stream) {
    (void)n_in; (void)out_size; (void)ws_size;

    const float* x    = (const float*)d_in[0];
    const int*   src0 = (const int*)d_in[1];
    const int*   dst0 = (const int*)d_in[2];
    const int*   src1 = (const int*)d_in[3];
    const int*   dst1 = (const int*)d_in[4];
    const int*   src2 = (const int*)d_in[5];
    const int*   dst2 = (const int*)d_in[6];
    const float* W0s  = (const float*)d_in[7];
    const float* W0n  = (const float*)d_in[8];
    const float* b0   = (const float*)d_in[9];
    const float* W1s  = (const float*)d_in[10];
    const float* W1n  = (const float*)d_in[11];
    const float* b1   = (const float*)d_in[12];
    const float* W2s  = (const float*)d_in[13];
    const float* W2n  = (const float*)d_in[14];
    const float* b2   = (const float*)d_in[15];

    const int E0 = in_sizes[1], E1 = in_sizes[3], E2 = in_sizes[5];
    const int N1 = 33792, N2 = 3072, N3 = 512;
    const int F0 = 256, H = 1024, NC = 47;
    const int NCpad = ((NC + BN - 1) / BN) * BN;   // 128

    char* ws = (char*)d_ws;
    size_t off = 0;
    auto carve = [&](size_t bytes) {
        void* p = ws + off;
        off += (bytes + 255) & ~(size_t)255;
        return p;
    };
    // bf16 activations / means
    __bf16* xb   = (__bf16*)carve((size_t)N1 * F0 * 2);
    __bf16* mb0  = (__bf16*)carve((size_t)N1 * F0 * 2);
    __bf16* h1   = (__bf16*)carve((size_t)N1 * H  * 2);
    __bf16* mb1  = (__bf16*)carve((size_t)N2 * H  * 2);
    __bf16* h2   = (__bf16*)carve((size_t)N2 * H  * 2);
    __bf16* mb2  = (__bf16*)carve((size_t)N3 * H  * 2);
    // fp32 aggregation scratch
    float* agg0 = (float*)carve((size_t)N1 * F0 * 4);
    float* agg1 = (float*)carve((size_t)N2 * H  * 4);
    float* agg2 = (float*)carve((size_t)N3 * H  * 4);
    float* deg0 = (float*)carve((size_t)N1 * 4);
    float* deg1 = (float*)carve((size_t)N2 * 4);
    float* deg2 = (float*)carve((size_t)N3 * 4);
    // transposed bf16 weights (rows padded to BN multiple)
    __bf16* WT0s = (__bf16*)carve((size_t)H * F0 * 2);
    __bf16* WT0n = (__bf16*)carve((size_t)H * F0 * 2);
    __bf16* WT1s = (__bf16*)carve((size_t)H * H * 2);
    __bf16* WT1n = (__bf16*)carve((size_t)H * H * 2);
    __bf16* WT2s = (__bf16*)carve((size_t)NCpad * H * 2);
    __bf16* WT2n = (__bf16*)carve((size_t)NCpad * H * 2);

    const int ZT = 256;
    auto zgrid = [&](long n) { return dim3((unsigned)((n + ZT - 1) / ZT)); };

    // ---- one-shot conversions (deterministic, every call)
    f32_to_bf16<<<zgrid((long)N1 * F0), ZT, 0, stream>>>(x, xb, (long)N1 * F0);
    wt_transpose_bf16<<<dim3(H), 256, 0, stream>>>(W0s, WT0s, F0, H);
    wt_transpose_bf16<<<dim3(H), 256, 0, stream>>>(W0n, WT0n, F0, H);
    wt_transpose_bf16<<<dim3(H), 256, 0, stream>>>(W1s, WT1s, H, H);
    wt_transpose_bf16<<<dim3(H), 256, 0, stream>>>(W1n, WT1n, H, H);
    wt_transpose_bf16<<<dim3(NCpad), 256, 0, stream>>>(W2s, WT2s, H, NC);
    wt_transpose_bf16<<<dim3(NCpad), 256, 0, stream>>>(W2n, WT2n, H, NC);

    // ---------------- Layer 0: [N0,256] -> [N1,1024], ReLU ----------------
    zero_f32<<<zgrid((long)N1 * F0), ZT, 0, stream>>>(agg0, (long)N1 * F0);
    zero_f32<<<zgrid(N1), ZT, 0, stream>>>(deg0, N1);
    scatter_add_f32<<<dim3(E0), 256, 0, stream>>>(x, src0, dst0, agg0, deg0, E0, F0);
    normalize_mean_bf16<<<dim3(N1), 256, 0, stream>>>(agg0, deg0, mb0, F0);
    sage_gemm_bf16_wmma<<<dim3(H / BN, N1 / BM), 256, 0, stream>>>(
        xb, mb0, WT0s, WT0n, b0, h1, nullptr, N1, F0, H, 1);

    // ---------------- Layer 1: [N1,1024] -> [N2,1024], ReLU ---------------
    zero_f32<<<zgrid((long)N2 * H), ZT, 0, stream>>>(agg1, (long)N2 * H);
    zero_f32<<<zgrid(N2), ZT, 0, stream>>>(deg1, N2);
    scatter_add_bf16<<<dim3(E1), 256, 0, stream>>>(h1, src1, dst1, agg1, deg1, E1, H);
    normalize_mean_bf16<<<dim3(N2), 256, 0, stream>>>(agg1, deg1, mb1, H);
    sage_gemm_bf16_wmma<<<dim3(H / BN, N2 / BM), 256, 0, stream>>>(
        h1, mb1, WT1s, WT1n, b1, h2, nullptr, N2, H, H, 1);

    // ---------------- Layer 2: [N2,1024] -> [N3,47], linear ---------------
    zero_f32<<<zgrid((long)N3 * H), ZT, 0, stream>>>(agg2, (long)N3 * H);
    zero_f32<<<zgrid(N3), ZT, 0, stream>>>(deg2, N3);
    scatter_add_bf16<<<dim3(E2), 256, 0, stream>>>(h2, src2, dst2, agg2, deg2, E2, H);
    normalize_mean_bf16<<<dim3(N3), 256, 0, stream>>>(agg2, deg2, mb2, H);
    sage_gemm_bf16_wmma<<<dim3(NCpad / BN, N3 / BM), 256, 0, stream>>>(
        h2, mb2, WT2s, WT2n, b2, nullptr, (float*)d_out, N3, H, NC, 0);
}
